// TemporalSelfAttention_19361712570501
// MI455X (gfx1250) — matrix operand order repaired and checked by reading, not compiled, not run on hardware
//
#include <hip/hip_runtime.h>
#include <hip/hip_bf16.h>
#include <stdint.h>

typedef __bf16 bf16;
typedef bf16  v16bf  __attribute__((ext_vector_type(16)));
typedef bf16  bf16x8 __attribute__((ext_vector_type(8)));
typedef float v8f    __attribute__((ext_vector_type(8)));

#define QN    16
#define EDIM  256
#define NHD   8
#define NLV   4
#define NPT   4

// ---- dynamic LDS byte offsets ----
#define SM_STAGE 0u        // f32  [16][256]      16KB  (input staging, reused 3x)
#define SM_QBF   16384u    // bf16 [16][256]       8KB  (query, bf16)
#define SM_HBF   24576u    // bf16 [16][256]       8KB  (key_hist then value_hist, bf16)
#define SM_QH    32768u    // bf16 [16][256]       8KB  (q projection)
#define SM_KH    40960u    // bf16 [2][16][256]   16KB  (k projections, s=0/1)
#define SM_VH    57344u    // bf16 [2][16][256]   16KB  (v projections, s=0/1)
#define SM_OFF   73728u    // f32  [16][256]      16KB  (sampling offsets)
#define SM_ATTN  90112u    // f32  [2][16][8][16] 16KB  (scores -> attn weights)
#define SM_CTX   106496u   // bf16 [16][256]       8KB  (sampled context)
#define SM_RP    114688u   // f32  [16][4][2]     512B  (reference points)
#define SM_TOTAL 115200u

// ---------------- async copy helpers (gfx1250 async-to-LDS path) -------------
__device__ __forceinline__ void async_b128(uint32_t lds_addr, uint64_t gaddr) {
  asm volatile("global_load_async_to_lds_b128 %0, %1, off"
               :: "v"(lds_addr), "v"(gaddr) : "memory");
}
__device__ __forceinline__ void wait_async0() {
  asm volatile("s_wait_asynccnt 0" ::: "memory");
}
// copy 16x256 fp32 (16KB) global -> LDS, all 256 threads, 4 x b128 each
__device__ __forceinline__ void stage_16x256(const float* g, char* smem, int t) {
  uint64_t ga = (uint64_t)(uintptr_t)g + (uint64_t)t * 64u;
  uint32_t la = (uint32_t)(uintptr_t)(smem + SM_STAGE) + (uint32_t)t * 64u;
#pragma unroll
  for (int i = 0; i < 4; ++i) async_b128(la + 16u * i, ga + 16u * i);
}

// ---------------- WMMA fragment helpers --------------------------------------
// A (16x32 bf16) per documented layout: lane<16 -> row m=lane, K = {k0..k0+7, k0+16..k0+23}
//                                       lane>=16 -> row m=lane-16, K shifted by 8.
__device__ __forceinline__ v16bf load_a(const bf16* A, int k0, int lane) {
  int m = lane & 15, half = lane >> 4;
  const bf16* p = A + m * EDIM + k0 + 8 * half;
  bf16x8 lo = *(const bf16x8*)(p);
  bf16x8 hi = *(const bf16x8*)(p + 16);
  return __builtin_shufflevector(lo, hi, 0,1,2,3,4,5,6,7,8,9,10,11,12,13,14,15);
}
// B (32x16 bf16), K-contiguous source rows of length 256:
// lane<16 -> col n=n0+lane, K = {k0..k0+7, k0+16..k0+23} ; lane>=16 -> col n0+lane-16, K shifted by 8
__device__ __forceinline__ v16bf load_b(const bf16* Wt, int k0, int n0, int lane) {
  int n = n0 + (lane & 15), half = lane >> 4;
  const bf16* p = Wt + n * EDIM + k0 + 8 * half;
  bf16x8 lo = *(const bf16x8*)(p);
  bf16x8 hi = *(const bf16x8*)(p + 16);
  return __builtin_shufflevector(lo, hi, 0,1,2,3,4,5,6,7,8,9,10,11,12,13,14,15);
}
// C/D layout: VGPR r -> row (r + 8*(lane>>4)), col n0 + (lane&15)
__device__ __forceinline__ void store_d_bf16(bf16* dst, v8f c, int n0, int lane,
                                             const float* bias) {
  int n = n0 + (lane & 15), half = lane >> 4;
  float bb = bias[n];
#pragma unroll
  for (int r = 0; r < 8; ++r) dst[(r + 8 * half) * EDIM + n] = (bf16)(c[r] + bb);
}
__device__ __forceinline__ void store_d_f32(float* dst, v8f c, int n0, int lane,
                                            const float* bias) {
  int n = n0 + (lane & 15), half = lane >> 4;
  float bb = bias[n];
#pragma unroll
  for (int r = 0; r < 8; ++r) dst[(r + 8 * half) * EDIM + n] = c[r] + bb;
}

__device__ __forceinline__ v8f wmma_bf16(v16bf a, v16bf b, v8f c) {
  return __builtin_amdgcn_wmma_f32_16x16x32_bf16(false, a, false, b,
                                                 (short)0, c, false, false);
}

// ---------------- weight prep: fp32 [K][N] -> bf16 [N][K] (K-contiguous) -----
__global__ void tsa_prep_weights(const float* __restrict__ Wq,
                                 const float* __restrict__ Wk,
                                 const float* __restrict__ Wv,
                                 const float* __restrict__ Woff,
                                 const float* __restrict__ Wout,
                                 bf16* __restrict__ ws) {
  int idx = blockIdx.x * blockDim.x + threadIdx.x;
  if (idx >= 5 * EDIM * EDIM) return;
  int w = idx >> 16;
  int r = idx & 65535;
  int n = r >> 8, k = r & 255;
  const float* W = (w == 0) ? Wq : (w == 1) ? Wk : (w == 2) ? Wv
                 : (w == 3) ? Woff : Wout;
  ws[idx] = (bf16)W[k * EDIM + n];   // ws[w][n][k]
}

// ---------------- fused temporal deformable attention ------------------------
__global__ void __launch_bounds__(256)
tsa_kernel(const float* __restrict__ query,
           const float* __restrict__ key_hist,
           const float* __restrict__ value_hist,
           const float* __restrict__ refpts,
           const float* __restrict__ bq, const float* __restrict__ bk,
           const float* __restrict__ bv, const float* __restrict__ boff,
           const float* __restrict__ bout,
           const bf16* __restrict__ ws,
           float* __restrict__ out) {
  extern __shared__ char smem[];
  float* stage = (float*)(smem + SM_STAGE);
  bf16*  qbf   = (bf16*)(smem + SM_QBF);
  bf16*  hbf   = (bf16*)(smem + SM_HBF);
  bf16*  qh    = (bf16*)(smem + SM_QH);
  bf16*  kh    = (bf16*)(smem + SM_KH);    // [2][16][256]
  bf16*  vh    = (bf16*)(smem + SM_VH);    // [2][16][256]
  float* offs  = (float*)(smem + SM_OFF);
  float* attn  = (float*)(smem + SM_ATTN); // [2][16][8][16]
  bf16*  ctxb  = (bf16*)(smem + SM_CTX);
  float* rpl   = (float*)(smem + SM_RP);   // [16][4][2]

  const int t = threadIdx.x, lane = t & 31;
  // wave index is uniform across the wave: force it scalar so all job
  // dispatch below is SGPR control flow (keeps EXEC untouched around WMMA,
  // enables saddr-based weight addressing instead of v_cndmask chains).
  const int wave = __builtin_amdgcn_readfirstlane(t >> 5);
  const int b = blockIdx.x;
  const float* qg = query      + (size_t)b * QN * EDIM;
  const float* kg = key_hist   + (size_t)b * QN * EDIM;
  const float* vg = value_hist + (size_t)b * QN * EDIM;
  const float* rg = refpts     + (size_t)b * QN * NLV * 2;

  const bf16* wsWq   = ws;
  const bf16* wsWk   = ws + 1 * 65536;
  const bf16* wsWv   = ws + 2 * 65536;
  const bf16* wsWoff = ws + 3 * 65536;
  const bf16* wsWout = ws + 4 * 65536;

  // warm L2 weight lines (global_prefetch_b8)
  __builtin_prefetch(wsWq  + t * 256, 0, 3);
  __builtin_prefetch(wsWout + t * 256, 0, 3);

  // ---- P0: async-stage query (+ reference points), convert to bf16 ----------
  stage_16x256(qg, smem, t);
  if (t < 32) {
    async_b128((uint32_t)(uintptr_t)(smem + SM_RP) + 16u * t,
               (uint64_t)(uintptr_t)rg + 16u * t);
  }
  wait_async0();
  __syncthreads();
#pragma unroll
  for (int i = 0; i < 16; ++i) qbf[t * 16 + i] = (bf16)stage[t * 16 + i];
  __syncthreads();

  // ---- P1: async-stage key_hist, convert to bf16 ----------------------------
  stage_16x256(kg, smem, t);
  wait_async0();
  __syncthreads();
#pragma unroll
  for (int i = 0; i < 16; ++i) hbf[t * 16 + i] = (bf16)stage[t * 16 + i];
  __syncthreads();

  // kick value_hist staging; it overlaps the projection WMMAs below
  stage_16x256(vg, smem, t);

  // ---- P2: projections: qh, kh0, kh1, off, vh0  (80 tiles, 10 per wave) -----
  for (int j = 0; j < 10; ++j) {
    int T = wave * 10 + j;            // scalar
    int job = T >> 4, n0 = (T & 15) * 16;
    const bf16* Asrc = (job == 2) ? hbf : qbf;
    const bf16* Wt = (job == 0) ? wsWq : (job <= 2) ? wsWk
                   : (job == 3) ? wsWoff : wsWv;
    v8f c = {};
#pragma unroll
    for (int kk = 0; kk < 8; ++kk) {
      v16bf a  = load_a(Asrc, kk * 32, lane);
      v16bf bb = load_b(Wt,   kk * 32, n0, lane);
      c = wmma_bf16(a, bb, c);
    }
    if      (job == 0) store_d_bf16(qh,        c, n0, lane, bq);
    else if (job == 1) store_d_bf16(kh,        c, n0, lane, bk);
    else if (job == 2) store_d_bf16(kh + 4096, c, n0, lane, bk);
    else if (job == 3) store_d_f32 (offs,      c, n0, lane, boff);
    else               store_d_bf16(vh,        c, n0, lane, bv);
  }
  wait_async0();
  __syncthreads();
#pragma unroll
  for (int i = 0; i < 16; ++i) hbf[t * 16 + i] = (bf16)stage[t * 16 + i];
  __syncthreads();

  // ---- P4: vh1 = value_hist @ Wv  (16 tiles, 2 per wave) --------------------
  for (int j = 0; j < 2; ++j) {
    int n0 = (wave * 2 + j) * 16;     // scalar
    v8f c = {};
#pragma unroll
    for (int kk = 0; kk < 8; ++kk) {
      v16bf a  = load_a(hbf,  kk * 32, lane);
      v16bf bb = load_b(wsWv, kk * 32, n0, lane);
      c = wmma_bf16(a, bb, c);
    }
    store_d_bf16(vh + 4096, c, n0, lane, bv);
  }
  __syncthreads();

  // ---- P5: attention scores  (16 WMMAs: s in {0,1}, h in {0..7}) -----------
  for (int j = 0; j < 2; ++j) {
    int sh = wave * 2 + j;            // scalar
    int s = sh >> 3, h = sh & 7;
    v16bf a  = load_a(qh, h * 32, lane);
    v16bf bb = load_b(kh + s * 4096, h * 32, 0, lane);   // B[k][n] = kh[s][n][h*32+k]
    v8f c = {};
    c = wmma_bf16(a, bb, c);
    int n = lane & 15, half = lane >> 4;
#pragma unroll
    for (int r = 0; r < 8; ++r) {
      int row = r + 8 * half;
      attn[((s * QN + row) * NHD + h) * 16 + n] = c[r];
    }
  }
  __syncthreads();

  // softmax: one (s,q,h) row per thread (2*16*8 = 256)
  {
    int s = t >> 7, q = (t >> 3) & 15, h = t & 7;
    float* row = attn + ((s * QN + q) * NHD + h) * 16;
    float mx = row[0];
#pragma unroll
    for (int i = 1; i < 16; ++i) mx = fmaxf(mx, row[i]);
    float sum = 0.f;
    float e[16];
#pragma unroll
    for (int i = 0; i < 16; ++i) { e[i] = __expf(row[i] - mx); sum += e[i]; }
    float inv = 1.f / sum;
#pragma unroll
    for (int i = 0; i < 16; ++i) row[i] = e[i] * inv;
  }
  __syncthreads();

  // ---- P6: bilinear sampling on 2x2 levels (zeros padding), mean over s -----
  {
    int q = t >> 4, h = (t >> 1) & 7, dh = t & 1;
    const float* at0 = attn + ((0 * QN + q) * NHD + h) * 16;
    const float* at1 = attn + ((1 * QN + q) * NHD + h) * 16;
    const bf16* v0 = vh;
    const bf16* v1 = vh + 4096;
    float acc0[16], acc1[16];
#pragma unroll
    for (int d = 0; d < 16; ++d) { acc0[d] = 0.f; acc1[d] = 0.f; }
    for (int l = 0; l < NLV; ++l) {
      float rx = rpl[(q * NLV + l) * 2 + 0];
      float ry = rpl[(q * NLV + l) * 2 + 1];
      for (int p = 0; p < NPT; ++p) {
        int oc = ((h * NLV + l) * NPT + p) * 2;
        float ox = offs[q * EDIM + oc], oy = offs[q * EDIM + oc + 1];
        // loc = rp + off/(W,H); grid = 2*loc-1; gx = (gridx+1)*W/2 - 0.5 (W=H=2)
        float gx = 2.f * (rx + 0.5f * ox) - 0.5f;
        float gy = 2.f * (ry + 0.5f * oy) - 0.5f;
        float x0 = floorf(gx), y0 = floorf(gy);
        float wx1 = gx - x0, wx0 = 1.f - wx1;
        float wy1 = gy - y0, wy0 = 1.f - wy1;
        int ix0 = (int)x0, iy0 = (int)y0;
        float aw0 = at0[l * NPT + p], aw1 = at1[l * NPT + p];
#pragma unroll
        for (int cc = 0; cc < 4; ++cc) {
          int xi = ix0 + (cc & 1), yi = iy0 + (cc >> 1);
          if (xi >= 0 && xi < 2 && yi >= 0 && yi < 2) {
            float wgt = ((cc & 1) ? wx1 : wx0) * ((cc >> 1) ? wy1 : wy0);
            int row = 4 * l + yi * 2 + xi;   // value row for this level pixel
            const bf16* p0 = v0 + row * EDIM + h * 32 + dh * 16;
            const bf16* p1 = v1 + row * EDIM + h * 32 + dh * 16;
            float w0 = aw0 * wgt, w1 = aw1 * wgt;
#pragma unroll
            for (int d = 0; d < 16; ++d) {
              acc0[d] += w0 * (float)p0[d];
              acc1[d] += w1 * (float)p1[d];
            }
          }
        }
      }
    }
    bf16* cd = ctxb + q * EDIM + h * 32 + dh * 16;
#pragma unroll
    for (int d = 0; d < 16; ++d) cd[d] = (bf16)(0.5f * (acc0[d] + acc1[d]));
  }
  __syncthreads();

  // ---- P7: out = ctx @ Wout + bout + residual(query) ------------------------
  for (int j = 0; j < 2; ++j) {
    int n0 = (wave * 2 + j) * 16;     // scalar
    v8f c = {};
#pragma unroll
    for (int kk = 0; kk < 8; ++kk) {
      v16bf a  = load_a(ctxb,   kk * 32, lane);
      v16bf bb = load_b(wsWout, kk * 32, n0, lane);
      c = wmma_bf16(a, bb, c);
    }
    int n = n0 + (lane & 15), half = lane >> 4;
    float bo = bout[n];
#pragma unroll
    for (int r = 0; r < 8; ++r) {
      int row = r + 8 * half;
      out[(size_t)b * QN * EDIM + row * EDIM + n] =
          c[r] + bo + qg[row * EDIM + n];
    }
  }
}

// ---------------- host launch -------------------------------------------------
extern "C" void kernel_launch(void* const* d_in, const int* in_sizes, int n_in,
                              void* d_out, int out_size, void* d_ws, size_t ws_size,
                              hipStream_t stream) {
  const float* query      = (const float*)d_in[0];
  const float* key_hist   = (const float*)d_in[1];
  const float* value_hist = (const float*)d_in[2];
  const float* refpts     = (const float*)d_in[3];
  // d_in[4] spatial_shapes: statically 2x2 per level
  const float* Wq   = (const float*)d_in[5];
  const float* bq   = (const float*)d_in[6];
  const float* Wk   = (const float*)d_in[7];
  const float* bk   = (const float*)d_in[8];
  const float* Wv   = (const float*)d_in[9];
  const float* bv   = (const float*)d_in[10];
  const float* Woff = (const float*)d_in[11];
  const float* boff = (const float*)d_in[12];
  const float* Wout = (const float*)d_in[13];
  const float* bout = (const float*)d_in[14];

  bf16* ws = (bf16*)d_ws;
  const int nW = 5 * EDIM * EDIM;
  tsa_prep_weights<<<(nW + 255) / 256, 256, 0, stream>>>(Wq, Wk, Wv, Woff, Wout, ws);

  int bs = in_sizes[0] / (QN * EDIM);   // 2048
  tsa_kernel<<<bs, 256, SM_TOTAL, stream>>>(query, key_hist, value_hist, refpts,
                                            bq, bk, bv, boff, bout, ws,
                                            (float*)d_out);
}
